// kMeanCluster_66348654788813
// MI455X (gfx1250) — compile-verified, matrix operand order checked by reading, not run on hardware
//
#include <hip/hip_runtime.h>
#include <hip/hip_bf16.h>
#include <stdint.h>

// ---------------------------------------------------------------------------
// CDNA5 WMMA (wave32): v_wmma_f32_16x16x32_bf16
//   A: 16x32 bf16  -> 16 bf16 per lane (8 VGPRs)
//   B: 32x16 bf16  -> 16 bf16 per lane (8 VGPRs), "column-major" (lane = col)
//   C/D: 16x16 f32 -> 8  f32  per lane
// ---------------------------------------------------------------------------
typedef __attribute__((ext_vector_type(16))) __bf16 v16bf;
typedef __attribute__((ext_vector_type(8)))  __bf16 v8bf;
typedef __attribute__((ext_vector_type(4)))  __bf16 v4bf;
typedef __attribute__((ext_vector_type(8)))  float  v8f;
typedef __attribute__((ext_vector_type(4)))  float  v4f;
typedef __attribute__((ext_vector_type(4)))  int    v4i;

// Problem constants (B,S,D,K) = (16,4096,256,1024)
#define NROWS (16 * 4096)   // 65536 feature vectors
#define DDIM  256
#define KCENT 1024

// gfx1250 async global->LDS copy path (ASYNCcnt-tracked), with fallback.
#if __has_builtin(__builtin_amdgcn_global_load_async_to_lds_b128) && \
    __has_builtin(__builtin_amdgcn_s_wait_asynccnt)
#define HAS_ASYNC_LDS 1
#else
#define HAS_ASYNC_LDS 0
#endif

#if HAS_ASYNC_LDS
// Builtin signature (from clang diagnostic): (int4 AS1*, int4 AS3*, Imm, Imm).
// AS1 pointers are 64-bit; AS3 pointers are 32-bit (LDS offset = low 32 bits
// of the flat address per the ISA aperture rule), so route casts through
// integers of the matching width.
__device__ __forceinline__ void async_cp16(const __bf16* g, __bf16* l) {
    v4i __attribute__((address_space(1)))* gp =
        reinterpret_cast<v4i __attribute__((address_space(1)))*>(
            reinterpret_cast<uintptr_t>(g));
    v4i __attribute__((address_space(3)))* lp =
        reinterpret_cast<v4i __attribute__((address_space(3)))*>(
            (unsigned)reinterpret_cast<uintptr_t>(l));
    __builtin_amdgcn_global_load_async_to_lds_b128(gp, lp, /*offset=*/0,
                                                   /*cpol=*/0);
}
#endif

// ---------------------------------------------------------------------------
// Exact fp32 squared row norms.
// ---------------------------------------------------------------------------
__global__ __launch_bounds__(256) void kmc_rownorms(const float* __restrict__ X,
                                                    float* __restrict__ out,
                                                    int rows) {
    int r = blockIdx.x * blockDim.x + threadIdx.x;
    if (r >= rows) return;
    const v4f* p = (const v4f*)(X + (size_t)r * DDIM);
    float s = 0.0f;
#pragma unroll 8
    for (int i = 0; i < DDIM / 4; ++i) {
        v4f v = p[i];
        s += v[0] * v[0] + v[1] * v[1] + v[2] * v[2] + v[3] * v[3];
    }
    out[r] = s;
}

// ---------------------------------------------------------------------------
// One-time fp32 -> bf16 hi/lo split of the centroid matrix (1024x256).
// ---------------------------------------------------------------------------
__global__ __launch_bounds__(256) void kmc_split(const float* __restrict__ src,
                                                 __bf16* __restrict__ hi,
                                                 __bf16* __restrict__ lo,
                                                 int n4) {
    int i = blockIdx.x * blockDim.x + threadIdx.x;
    if (i >= n4) return;
    v4f v = ((const v4f*)src)[i];
    v4bf h, l;
#pragma unroll
    for (int j = 0; j < 4; ++j) {
        __bf16 hv = (__bf16)v[j];
        h[j] = hv;
        l[j] = (__bf16)(v[j] - (float)hv);
    }
    ((v4bf*)hi)[i] = h;
    ((v4bf*)lo)[i] = l;
}

// On-the-fly hi/lo split for the A fragment (16 fp32 per lane per k-chunk).
__device__ __forceinline__ void load_split16(const float* __restrict__ p0,
                                             const float* __restrict__ p1,
                                             v16bf& hi, v16bf& lo) {
    v4f a = *(const v4f*)(p0);
    v4f b = *(const v4f*)(p0 + 4);
    v4f c = *(const v4f*)(p1);
    v4f d = *(const v4f*)(p1 + 4);
    float v[16] = {a[0], a[1], a[2], a[3], b[0], b[1], b[2], b[3],
                   c[0], c[1], c[2], c[3], d[0], d[1], d[2], d[3]};
#pragma unroll
    for (int i = 0; i < 16; ++i) {
        __bf16 h = (__bf16)v[i];
        hi[i] = h;
        lo[i] = (__bf16)(v[i] - (float)h);
    }
}

__device__ __forceinline__ v8f wmma_bf16(v16bf a, v16bf b, v8f c) {
    // 8 args: (neg_a, A, neg_b, B, c_mod, C, reuse_a, reuse_b)
    return __builtin_amdgcn_wmma_f32_16x16x32_bf16(false, a, false, b,
                                                   (short)0, c, false, false);
}

// Stage one (ns=wave, lane) B-fragment pair (hi/lo, 4x16B) into LDS.
__device__ __forceinline__ void stage_frag(const __bf16* __restrict__ chi_row,
                                           const __bf16* __restrict__ clo_row,
                                           int kb, int koff,
                                           __bf16* lhi, __bf16* llo) {
#if HAS_ASYNC_LDS
    async_cp16(chi_row + kb + koff,      lhi);
    async_cp16(chi_row + kb + 16 + koff, lhi + 8);
    async_cp16(clo_row + kb + koff,      llo);
    async_cp16(clo_row + kb + 16 + koff, llo + 8);
#else
    *(v8bf*)(lhi)     = *(const v8bf*)(chi_row + kb + koff);
    *(v8bf*)(lhi + 8) = *(const v8bf*)(chi_row + kb + 16 + koff);
    *(v8bf*)(llo)     = *(const v8bf*)(clo_row + kb + koff);
    *(v8bf*)(llo + 8) = *(const v8bf*)(clo_row + kb + 16 + koff);
#endif
}

__device__ __forceinline__ void wait_stage() {
#if HAS_ASYNC_LDS
    __builtin_amdgcn_s_wait_asynccnt(0);
#endif
    __syncthreads();
}

// ---------------------------------------------------------------------------
// Main kernel. Block = 256 threads = 8 waves; block computes 128(M) x 128(N).
// Per k-chunk the block stages the shared 8 B-fragment pairs (hi/lo) in a
// double-buffered LDS (2 x 16 KB): iteration i waits on buffer i&1, launches
// the async fill of buffer (i+1)&1, then runs 24 WMMAs from buffer i&1 while
// the next chunk's copies are in flight.
//   dist[m][n] = x2[m] + c2[n] - 2 * (Ahi*Bhi + Ahi*Blo + Alo*Bhi)
// ---------------------------------------------------------------------------
__global__ __launch_bounds__(256) void kmc_dist_wmma(
    const float*  __restrict__ X,    // (NROWS, D) fp32
    const __bf16* __restrict__ Chi,  // (KCENT, D) bf16 hi
    const __bf16* __restrict__ Clo,  // (KCENT, D) bf16 lo
    const float*  __restrict__ x2,   // (NROWS)
    const float*  __restrict__ c2,   // (KCENT)
    float* __restrict__ out) {       // (NROWS, KCENT)

    // LDS: [buf][ns][hilo][lane][16 bf16]  = 2 * 16 KB
    __shared__ __bf16 lds[2][8 * 2 * 32 * 16];

    const int t     = threadIdx.x;
    const int wave  = t >> 5;
    const int lane  = t & 31;
    const int laneM = lane & 15;
    const int half  = lane >> 4;         // 0: K offset +0, 1: K offset +8
    const int koff  = 8 * half;

    const int mblk  = blockIdx.x >> 3;   // NROWS/128 = 512 row blocks
    const int ntile = blockIdx.x & 7;    // KCENT/128 = 8 col tiles
    const int row0  = mblk * 128 + wave * 16;
    const int col0  = ntile * 128;

    // Fill role: thread t owns fragment slot (ns = wave, lane).
    const size_t brow = (size_t)(col0 + wave * 16 + laneM) * DDIM;
    const __bf16* chi_row = Chi + brow;
    const __bf16* clo_row = Clo + brow;
    const int slot_hi = ((wave * 2 + 0) * 32 + lane) * 16;
    const int slot_lo = ((wave * 2 + 1) * 32 + lane) * 16;

    v8f z = {0.f, 0.f, 0.f, 0.f, 0.f, 0.f, 0.f, 0.f};
    v8f acc0 = z, acc1 = z, acc2 = z, acc3 = z;
    v8f acc4 = z, acc5 = z, acc6 = z, acc7 = z;
    v8f* acc[8] = {&acc0, &acc1, &acc2, &acc3, &acc4, &acc5, &acc6, &acc7};

    const float* arow = X + (size_t)(row0 + laneM) * DDIM;

    // Prologue: stage first k-chunk into buffer 0.
    stage_frag(chi_row, clo_row, 0, koff, &lds[0][slot_hi], &lds[0][slot_lo]);

#pragma unroll 1
    for (int i = 0; i < DDIM / 32; ++i) {
        const int kb  = i * 32;
        const int cur = i & 1;

        // Wait for buffer `cur` (async copies issued last iteration) and make
        // it visible block-wide.
        wait_stage();

        // Kick off the next chunk's copies into the other buffer; they run
        // underneath this iteration's WMMAs.
        if (i + 1 < DDIM / 32) {
            const int nxt = (i + 1) & 1;
            stage_frag(chi_row, clo_row, kb + 32, koff,
                       &lds[nxt][slot_hi], &lds[nxt][slot_lo]);
        }

        // A fragment: split 16 fp32 -> bf16 hi/lo in registers.
        v16bf ahi, alo;
        load_split16(arow + kb + koff, arow + kb + 16 + koff, ahi, alo);

#pragma unroll
        for (int ns = 0; ns < 8; ++ns) {
            v16bf bhi = *(const v16bf*)&lds[cur][((ns * 2 + 0) * 32 + lane) * 16];
            v16bf blo = *(const v16bf*)&lds[cur][((ns * 2 + 1) * 32 + lane) * 16];
            v8f a = *acc[ns];
            a = wmma_bf16(ahi, bhi, a);   // hi*hi
            a = wmma_bf16(ahi, blo, a);   // hi*lo
            a = wmma_bf16(alo, bhi, a);   // lo*hi (lo*lo ~2^-16, dropped)
            *acc[ns] = a;
        }
    }

    // Epilogue. D layout: lane L -> N = (L&15); VGPR r -> M = r + 8*half.
    float xn[8];
#pragma unroll
    for (int r = 0; r < 8; ++r) xn[r] = x2[row0 + 8 * half + r];

#pragma unroll
    for (int ns = 0; ns < 8; ++ns) {
        const int col = col0 + ns * 16 + laneM;
        const float cn = c2[col];
        v8f a = *acc[ns];
#pragma unroll
        for (int r = 0; r < 8; ++r) {
            const size_t row = (size_t)(row0 + 8 * half + r);
            out[row * KCENT + col] = xn[r] + cn - 2.0f * a[r];
        }
    }
}

// ---------------------------------------------------------------------------
// Launch
// ---------------------------------------------------------------------------
extern "C" void kernel_launch(void* const* d_in, const int* in_sizes, int n_in,
                              void* d_out, int out_size, void* d_ws, size_t ws_size,
                              hipStream_t stream) {
    const float* X = (const float*)d_in[0];   // (16,4096,256) fp32
    const float* C = (const float*)d_in[1];   // (1,1024,256) fp32
    float* out = (float*)d_out;               // (16,4096,1024) fp32

    // Workspace layout (~1.3 MB total):
    float*  x2  = (float*)d_ws;                       // NROWS floats
    float*  c2  = x2 + NROWS;                         // KCENT floats
    __bf16* Chi = (__bf16*)(c2 + KCENT);              // KCENT*DDIM bf16
    __bf16* Clo = Chi + (size_t)KCENT * DDIM;         // KCENT*DDIM bf16

    kmc_rownorms<<<NROWS / 256, 256, 0, stream>>>(X, x2, NROWS);
    kmc_rownorms<<<(KCENT + 255) / 256, 256, 0, stream>>>(C, c2, KCENT);

    const int n4 = (KCENT * DDIM) / 4;                // 65536 float4 groups
    kmc_split<<<n4 / 256, 256, 0, stream>>>(C, Chi, Clo, n4);

    // Block = 128x128 output tile: (NROWS/128) * (KCENT/128) = 4096 blocks.
    const int blocks = (NROWS / 128) * (KCENT / 128);
    kmc_dist_wmma<<<blocks, 256, 0, stream>>>(X, Chi, Clo, x2, c2, out);
}
